// DSMoE_71193377898704
// MI455X (gfx1250) — matrix operand order repaired and checked by reading, not compiled, not run on hardware
//
#include <hip/hip_runtime.h>

#define C_DIM 256
#define H_DIM 1024
#define E_NUM 32
#define G_NUM 31
#define M_TILE 32

// dynamic LDS layout (bytes)
#define XS_BYTES (M_TILE * 264 * 2)                  // x tile bf16 [32][264]
#define HS_BYTES (M_TILE * 1032 * 2)                 // h tile bf16 [32][1032]
#define YS_BYTES (M_TILE * 264 * 4)                  // y tile f32 [32][264]
#define SMEM_BYTES (XS_BYTES + HS_BYTES + YS_BYTES)  // 116736 < 320KB/WGP

typedef __attribute__((ext_vector_type(16))) __bf16 bf16x16;
typedef __attribute__((ext_vector_type(8)))  float  floatx8;
typedef __attribute__((ext_vector_type(8)))  unsigned int uintx8;

union FragCvt { uintx8 u; bf16x16 b; };

__device__ __forceinline__ unsigned short f2bf(float f) {
  unsigned int u = __float_as_uint(f);
  u += 0x7FFFu + ((u >> 16) & 1u);          // round-to-nearest-even
  return (unsigned short)(u >> 16);
}

__device__ __forceinline__ float gelu_exact(float x) {
  return 0.5f * x * (1.0f + erff(x * 0.70710678118654752f));
}

// A fragment (16x32 bf16, ISA 7.12.2): lane L holds row M=L&15;
// K = (L>=16? 8:0)+{0..7} and 16+(L>=16? 8:0)+{0..7} -> two ds_load_b128.
__device__ __forceinline__ bf16x16 load_a_frag_lds(const unsigned short* base,
                                                   int stride, int k0, int lane) {
  int m = lane & 15, half = lane >> 4;
  const uint4* p0 = (const uint4*)(base + m * stride + k0 + half * 8);
  const uint4* p1 = (const uint4*)(base + m * stride + k0 + 16 + half * 8);
  uint4 a = *p0, b = *p1;
  FragCvt f;
  f.u[0] = a.x; f.u[1] = a.y; f.u[2] = a.z; f.u[3] = a.w;
  f.u[4] = b.x; f.u[5] = b.y; f.u[6] = b.z; f.u[7] = b.w;
  return f.b;
}

// B fragment from pre-packed weights: 1024B tile, lane L owns 32 contiguous
// bytes -> two coalesced global_load_b128, zero conversion VALU in the k-loop.
__device__ __forceinline__ bf16x16 load_b_frag_packed(const unsigned short* tile,
                                                      int lane) {
  const uint4* p = (const uint4*)(tile + lane * 16);
  uint4 a = p[0], b = p[1];
  FragCvt f;
  f.u[0] = a.x; f.u[1] = a.y; f.u[2] = a.z; f.u[3] = a.w;
  f.u[4] = b.x; f.u[5] = b.y; f.u[6] = b.z; f.u[7] = b.w;
  return f.b;
}

// One-pass fp32 -> bf16 weight pre-pack into B-fragment swizzle.
// Tile (kt,nt) covers K=[kt*32,+32), N=[nt*16,+16); 64 uint4 per tile.
__global__ __launch_bounds__(256)
void pack_weights_kernel(const float* __restrict__ W, unsigned short* __restrict__ swz,
                         int K, int N, long total_u4) {
  long d4 = (long)blockIdx.x * blockDim.x + threadIdx.x;
  if (d4 >= total_u4) return;
  long per_e = (long)K * N / 8;            // uint4 per expert
  int e = (int)(d4 / per_e);
  long r = d4 - (long)e * per_e;
  int ntile_n = N >> 4;
  long t = r >> 6;                         // tile index within expert
  int r4 = (int)(r & 63);
  int lane = r4 >> 1;
  int vq = (r4 & 1) << 2;                  // v base 0 or 4 (4 uints = 8 bf16)
  int kt = (int)(t / ntile_n);
  int nt = (int)(t - (long)kt * ntile_n);
  int k = kt * 32 + ((lane >> 4) << 3) + 4 * vq;
  int n = nt * 16 + (lane & 15);
  const float* src = W + (long)e * K * N + (long)k * N + n;
  uint4 out;
  unsigned int* po = &out.x;
#pragma unroll
  for (int v = 0; v < 4; ++v) {
    unsigned int lo = f2bf(src[0]);
    unsigned int hi = f2bf(src[N]);
    po[v] = lo | (hi << 16);
    src += 2 * (long)N;
  }
  ((uint4*)swz)[d4] = out;
}

__global__ void zero_cnt_kernel(int* __restrict__ cnt) {
  if (threadIdx.x < E_NUM) cnt[threadIdx.x] = 0;
}

// One wave per token: logits, softmax, top-3, rw row, expert list append.
__global__ __launch_bounds__(256)
void gate_kernel(const float* __restrict__ x, const float* __restrict__ Wg,
                 const float* __restrict__ bg, const float* __restrict__ eb,
                 float* __restrict__ rw, int* __restrict__ cnt,
                 int* __restrict__ list, int n) {
  int wv = threadIdx.x >> 5, lane = threadIdx.x & 31;
  int token = blockIdx.x * 8 + wv;
  if (token >= n) return;
  const float* xr = x + (long)token * C_DIM;

  float l = -1e30f;
  if (lane < G_NUM) {
    float acc = bg[lane];
    for (int k = 0; k < C_DIM; ++k) acc += xr[k] * Wg[k * G_NUM + lane];
    l = acc;
  }
  float m = l;
  for (int off = 16; off > 0; off >>= 1) m = fmaxf(m, __shfl_xor(m, off, 32));
  float ev = (lane < G_NUM) ? expf(l - m) : 0.f;
  float s = ev;
  for (int off = 16; off > 0; off >>= 1) s += __shfl_xor(s, off, 32);
  float g = ev / s;
  float v = (lane < G_NUM) ? (g + eb[lane]) : -1e30f;

  float gv[3]; int gi[3];
#pragma unroll
  for (int t = 0; t < 3; ++t) {
    float bv = v; int bi = lane;
    for (int off = 16; off > 0; off >>= 1) {
      float ov = __shfl_xor(bv, off, 32);
      int   oi = __shfl_xor(bi, off, 32);
      if (ov > bv || (ov == bv && oi < bi)) { bv = ov; bi = oi; }
    }
    gv[t] = bv; gi[t] = bi;
    if (lane == bi) v = -1e30f;
  }
  float s3 = gv[0] + gv[1] + gv[2];
  float w0 = gv[0] / s3 * 0.75f;
  float w1 = gv[1] / s3 * 0.75f;
  float w2 = gv[2] / s3 * 0.75f;

  float val = 0.f;
  if (lane == 0) val = 0.25f;
  else {
    int e = lane - 1;
    if      (e == gi[0]) val = w0;
    else if (e == gi[1]) val = w1;
    else if (e == gi[2]) val = w2;
  }
  rw[(long)token * E_NUM + lane] = val;

  if (lane < 4) {
    int e = (lane == 0) ? 0 : gi[lane - 1] + 1;
    int pos = atomicAdd(&cnt[e], 1);
    list[e * n + pos] = token * 4 + lane;   // pack token|slot
  }
}

__global__ __launch_bounds__(256)
void expert_kernel(const float* __restrict__ x, const float* __restrict__ rf,
                   const unsigned short* __restrict__ swzFc,
                   const float* __restrict__ bfc,
                   const unsigned short* __restrict__ swzP,
                   const float* __restrict__ bproj,
                   const float* __restrict__ cptr, const float* __restrict__ rw,
                   const int* __restrict__ cnt, const int* __restrict__ list,
                   float* __restrict__ cbuf, int n) {
  extern __shared__ __align__(16) char smem[];
  unsigned short* xs = (unsigned short*)smem;                       // [32][264]
  unsigned short* hs = (unsigned short*)(smem + XS_BYTES);          // [32][1032]
  float*          ys = (float*)(smem + XS_BYTES + HS_BYTES);        // [32][264]
  __shared__ int rowtok[M_TILE]; __shared__ float roww[M_TILE];
  __shared__ int rowslot[M_TILE];

  int e = blockIdx.y;
  int tile = blockIdx.x;
  int count = cnt[e];
  if (tile * M_TILE >= count) return;       // uniform early-out for empty tiles
  int tid = threadIdx.x;
  int lane = tid & 31, wv = tid >> 5;
  int validn = count - tile * M_TILE; if (validn > M_TILE) validn = M_TILE;

  if (tid < M_TILE) {
    int i = tile * M_TILE + tid;
    int packed = list[e * n + ((i < count) ? i : tile * M_TILE)];  // pad dup
    rowtok[tid]  = packed >> 2;
    rowslot[tid] = packed & 3;
    roww[tid] = (i < count) ? rw[(long)(packed >> 2) * E_NUM + e] : 0.f;
  }
  __syncthreads();

  // gather 32 token rows of x into LDS as bf16
  for (int it = 0; it < M_TILE; ++it) {
    int idx = tid + 256 * it;
    int r = idx >> 8, col = idx & 255;
    xs[r * 264 + col] = f2bf(x[(long)rowtok[r] * C_DIM + col]);
  }
  __syncthreads();

  const unsigned short* swzFc_e = swzFc + (long)e * C_DIM * H_DIM;
  const unsigned short* swzP_e  = swzP  + (long)e * H_DIM * C_DIM;
  const floatx8 vzero = {0.f,0.f,0.f,0.f,0.f,0.f,0.f,0.f};
  int half = lane >> 4, nl = lane & 15;

  // ---- stage 1: h = gelu(x @ Wfc + bfc) ----
  // Each B fragment feeds 2 WMMAs (M=32 -> 2 M-subtiles): 2x L2 reuse.
  // Two N-passes of 4 N-tiles keep accumulators at 8 x v8f = 64 VGPRs.
  {
    const int KT_STRIDE = (H_DIM / 16) * 512;  // ushorts per kt row of tiles
#pragma unroll 1
    for (int npass = 0; npass < 2; ++npass) {
      int hbase = npass * 512 + wv * 64;       // 4 N-tiles per pass per wave
      floatx8 acc0[4], acc1[4];
#pragma unroll
      for (int j = 0; j < 4; ++j) { acc0[j] = vzero; acc1[j] = vzero; }
      for (int k0 = 0; k0 < C_DIM; k0 += 32) {
        bf16x16 a0 = load_a_frag_lds(xs,             264, k0, lane);
        bf16x16 a1 = load_a_frag_lds(xs + 16 * 264,  264, k0, lane);
        const unsigned short* krow = swzFc_e + (long)(k0 >> 5) * KT_STRIDE
                                             + (hbase >> 4) * 512;
        if (k0 + 32 < C_DIM)
          __builtin_prefetch(krow + KT_STRIDE + lane * 16, 0, 3);
#pragma unroll
        for (int j = 0; j < 4; ++j) {
          bf16x16 b = load_b_frag_packed(krow + j * 512, lane);
          acc0[j] = __builtin_amdgcn_wmma_f32_16x16x32_bf16(
              false, a0, false, b, (short)0, acc0[j], false, false);
          acc1[j] = __builtin_amdgcn_wmma_f32_16x16x32_bf16(
              false, a1, false, b, (short)0, acc1[j], false, false);
        }
      }
#pragma unroll
      for (int j = 0; j < 4; ++j) {
        int N = hbase + 16 * j + nl;
        float bias = bfc[(long)e * H_DIM + N];
#pragma unroll
        for (int r = 0; r < 8; ++r) {
          int M0 = r + 8 * half;                 // C/D layout (ISA 7.12.2)
          hs[M0 * 1032 + N]        = f2bf(gelu_exact(acc0[r == r ? j : j][r] + bias));
          hs[(M0 + 16) * 1032 + N] = f2bf(gelu_exact(acc1[j][r] + bias));
        }
      }
    }
  }
  __syncthreads();

  // ---- stage 2: y = h @ Wproj + bproj; 2 N-tiles x 2 M-subtiles per wave ----
  {
    const int KT_STRIDE = (C_DIM / 16) * 512;
    int nb = wv * 32;
    floatx8 acc0[2], acc1[2];
    acc0[0] = vzero; acc0[1] = vzero; acc1[0] = vzero; acc1[1] = vzero;
    for (int k0 = 0; k0 < H_DIM; k0 += 32) {
      bf16x16 a0 = load_a_frag_lds(hs,              1032, k0, lane);
      bf16x16 a1 = load_a_frag_lds(hs + 16 * 1032,  1032, k0, lane);
      const unsigned short* krow = swzP_e + (long)(k0 >> 5) * KT_STRIDE
                                          + (nb >> 4) * 512;
      if (k0 + 32 < H_DIM)
        __builtin_prefetch(krow + KT_STRIDE + lane * 16, 0, 3);
#pragma unroll
      for (int j = 0; j < 2; ++j) {
        bf16x16 b = load_b_frag_packed(krow + j * 512, lane);
        acc0[j] = __builtin_amdgcn_wmma_f32_16x16x32_bf16(
            false, a0, false, b, (short)0, acc0[j], false, false);
        acc1[j] = __builtin_amdgcn_wmma_f32_16x16x32_bf16(
            false, a1, false, b, (short)0, acc1[j], false, false);
      }
    }
#pragma unroll
    for (int j = 0; j < 2; ++j) {
      int N = nb + 16 * j + nl;
      float bias = bproj[(long)e * C_DIM + N];
#pragma unroll
      for (int r = 0; r < 8; ++r) {
        ys[(r + 8 * half) * 264 + N]      = acc0[j][r] + bias;
        ys[(r + 8 * half + 16) * 264 + N] = acc1[j][r] + bias;
      }
    }
  }
  __syncthreads();

  // ---- epilogue: Poincare expmap + routing weight, 4 rows per wave ----
  float cc = cptr[0];
  float sqc = sqrtf(cc);
#pragma unroll 1
  for (int rr = 0; rr < 4; ++rr) {
    int m = wv * 4 + rr;
    int tok = rowtok[m]; float w = roww[m]; int slot = rowslot[m];
    const float* rfr = rf + (long)tok * C_DIM;
    float yv[8], rv[8];
    float sy2 = 0.f, sr2 = 0.f, sry = 0.f;
#pragma unroll
    for (int q = 0; q < 8; ++q) {
      int col = lane + 32 * q;                  // coalesced rf loads
      float yq = ys[m * 264 + col];
      float rq = rfr[col];
      yv[q] = yq; rv[q] = rq;
      sy2 += yq * yq; sr2 += rq * rq; sry += rq * yq;
    }
    for (int off = 16; off > 0; off >>= 1) {
      sy2 += __shfl_xor(sy2, off, 32);
      sr2 += __shfl_xor(sr2, off, 32);
      sry += __shfl_xor(sry, off, 32);
    }
    float sf  = 2.f / (1.f + cc * sr2);
    float vn  = sqrtf(sy2);
    float arg = sqrtf(cc * sf * 0.5f) * vn;
    float kk  = tanhf(arg) / (sqc * vn + 1e-30f);  // second = kk * y
    float xy  = kk * sry;
    float y2s = kk * kk * sy2;
    float alpha = 1.f + 2.f * cc * xy + cc * y2s;
    float beta  = 1.f - cc * sr2;
    float den   = 1.f + 2.f * cc * xy + cc * cc * sr2 * y2s;
    float scale = w / den;
    if (m < validn) {                            // never write pad rows
      float* dst = cbuf + ((long)tok * 4 + slot) * C_DIM;
#pragma unroll
      for (int q = 0; q < 8; ++q) {
        int col = lane + 32 * q;
        dst[col] = scale * (alpha * rv[q] + beta * kk * yv[q]);
      }
    }
  }
}

// deterministic reduction of the 4 slot contributions per token
__global__ void combine_kernel(const float* __restrict__ cbuf,
                               float* __restrict__ out, int total) {
  int i = blockIdx.x * blockDim.x + threadIdx.x;
  if (i >= total) return;
  int tok = i >> 8, col = i & 255;
  const float* p = cbuf + (long)tok * 4 * C_DIM + col;
  out[i] = p[0] + p[C_DIM] + p[2 * C_DIM] + p[3 * C_DIM];
}

extern "C" void kernel_launch(void* const* d_in, const int* in_sizes, int n_in,
                              void* d_out, int out_size, void* d_ws, size_t ws_size,
                              hipStream_t stream) {
  const float* x   = (const float*)d_in[0];
  const float* rfp = (const float*)d_in[1];
  const float* Wg  = (const float*)d_in[2];
  const float* bg  = (const float*)d_in[3];
  const float* eb  = (const float*)d_in[4];
  const float* Wfc = (const float*)d_in[5];
  const float* bfc = (const float*)d_in[6];
  const float* Wp  = (const float*)d_in[7];
  const float* bp  = (const float*)d_in[8];
  const float* c   = (const float*)d_in[9];
  (void)n_in; (void)out_size; (void)ws_size;

  int n = in_sizes[0] / C_DIM;                 // tokens
  float* out = (float*)d_out;                  // [n, C]
  float* rw  = out + (long)n * C_DIM;          // [n, E] second tuple output

  char* ws  = (char*)d_ws;
  int* cnt  = (int*)ws;                        // 32 ints
  int* list = (int*)(ws + 128);                // [E, n] packed token|slot
  size_t off = ((size_t)128 + (size_t)E_NUM * (size_t)n * 4 + 255) & ~(size_t)255;
  float* cbuf = (float*)(ws + off);            // [n, 4, C] contributions
  off = (off + (size_t)n * 4 * C_DIM * 4 + 255) & ~(size_t)255;
  unsigned short* swzFc = (unsigned short*)(ws + off);   // packed bf16 Wfc
  off += (size_t)E_NUM * C_DIM * H_DIM * 2;
  unsigned short* swzP  = (unsigned short*)(ws + off);   // packed bf16 Wproj

  long u4_fc = (long)E_NUM * C_DIM * H_DIM / 8;
  long u4_p  = (long)E_NUM * H_DIM * C_DIM / 8;

  static int smem_cfgd = 0;
  if (!smem_cfgd) {   // host-side attribute, idempotent, capture-safe
    hipFuncSetAttribute((const void*)expert_kernel,
                        hipFuncAttributeMaxDynamicSharedMemorySize, SMEM_BYTES);
    smem_cfgd = 1;
  }

  zero_cnt_kernel<<<1, 32, 0, stream>>>(cnt);
  pack_weights_kernel<<<(int)((u4_fc + 255) / 256), 256, 0, stream>>>(
      Wfc, swzFc, C_DIM, H_DIM, u4_fc);
  pack_weights_kernel<<<(int)((u4_p + 255) / 256), 256, 0, stream>>>(
      Wp, swzP, H_DIM, C_DIM, u4_p);
  gate_kernel<<<(n + 7) / 8, 256, 0, stream>>>(x, Wg, bg, eb, rw, cnt, list, n);
  dim3 grid((n + M_TILE - 1) / M_TILE, E_NUM);
  expert_kernel<<<grid, 256, SMEM_BYTES, stream>>>(x, rfp, swzFc, bfc, swzP, bp,
                                                   c, rw, cnt, list, cbuf, n);
  combine_kernel<<<(n * C_DIM + 255) / 256, 256, 0, stream>>>(cbuf, out, n * C_DIM);
}
